// SelfAttention_45311904973058
// MI455X (gfx1250) — matrix operand order repaired and checked by reading, not compiled, not run on hardware
//
#include <hip/hip_runtime.h>

#define BATCH 8
#define SLEN  2048
#define DDIM  512
#define SCALE 0.04419417382415922f  // 1/sqrt(512)

#define KSTR  520   // K tile LDS row stride (elems): 260 dwords -> 4-bank stagger
#define WSTR  40    // W tile LDS row stride (elems): 20 dwords  -> conflict-free
#define SSTR  36    // score LDS row stride (floats): 9 v4f      -> conflict-free
#define PSTR  40    // P tile LDS row stride (elems)

typedef __attribute__((ext_vector_type(16))) __bf16        v16bf;
typedef __attribute__((ext_vector_type(8)))  float         v8f;
typedef __attribute__((ext_vector_type(4)))  float         v4f;
typedef __attribute__((ext_vector_type(4)))  unsigned int  u32x4;
typedef __attribute__((ext_vector_type(2)))  unsigned int  u32x2;

union Frag {          // 16 bf16 per lane = one WMMA A or B operand slice
  u32x4 q[2];
  v16bf v;
};

__device__ __forceinline__ unsigned short f2bf(float f) {
  unsigned int u = __builtin_bit_cast(unsigned int, f);
  u += 0x7FFFu + ((u >> 16) & 1u);            // round-to-nearest-even
  return (unsigned short)(u >> 16);
}

__device__ __forceinline__ v8f wmma_bf16(v16bf a, v16bf b, v8f c) {
  return __builtin_amdgcn_wmma_f32_16x16x32_bf16(false, a, false, b, (short)0, c,
                                                 false, false);
}

// ---- CDNA5 async global->LDS copy (ASYNCcnt-tracked, no VGPR round-trip) ----
__device__ __forceinline__ unsigned lds_off(const void* p) {
  return (unsigned)(size_t)p;                 // low 32 bits = LDS offset
}
__device__ __forceinline__ void async_ld16(unsigned ldsoff, const void* gp) {
  asm volatile("global_load_async_to_lds_b128 %0, %1, off"
               :: "v"(ldsoff), "v"(gp) : "memory");
}
__device__ __forceinline__ void wait_async_all() {
  asm volatile("s_wait_asynccnt 0x0" ::: "memory");
}

// ---------------------------------------------------------------------------
// Stage 0a: x (f32) -> Xb (bf16)
// ---------------------------------------------------------------------------
__global__ __launch_bounds__(256) void cvt_x_bf16(const float* __restrict__ x,
                                                  unsigned short* __restrict__ xb) {
  const size_t i = ((size_t)blockIdx.x * 256 + threadIdx.x) * 4;
  v4f v = *(const v4f*)(x + i);
  u32x2 o;
  o.x = (unsigned int)f2bf(v[0]) | ((unsigned int)f2bf(v[1]) << 16);
  o.y = (unsigned int)f2bf(v[2]) | ((unsigned int)f2bf(v[3]) << 16);
  *(u32x2*)(xb + i) = o;
}

// ---------------------------------------------------------------------------
// Stage 0b: W [k][n] f32 -> Wt [n][k] bf16
// ---------------------------------------------------------------------------
__global__ __launch_bounds__(256) void cvt_w_t(const float* __restrict__ w,
                                               unsigned short* __restrict__ wt) {
  const int idx = blockIdx.x * 256 + threadIdx.x;
  const int k = idx >> 9;
  const int n = idx & 511;
  wt[(size_t)n * DDIM + k] = f2bf(w[(size_t)k * DDIM + n]);
}

// ---------------------------------------------------------------------------
// Stage 1: projection GEMM  dst = Xb @ W + b.  8-wave WG computes 128x128.
// W chunk tiles (128x32) double-buffered in LDS via async global->LDS copies;
// one barrier per chunk. A-fragments direct from global with prefetch.
// ---------------------------------------------------------------------------
__global__ __launch_bounds__(256) void qkv_proj(const unsigned short* __restrict__ Xb,
                                               const unsigned short* __restrict__ Wt,
                                               const float* __restrict__ bias,
                                               unsigned short* __restrict__ dst,
                                               const int transposed) {
  __shared__ __align__(16) unsigned short wlds[2][128 * WSTR];

  const int tid  = threadIdx.x;
  const int lane = tid & 31;
  const int wave = tid >> 5;
  const int nloc = lane & 15;
  const int hi   = lane >> 4;

  const int mrow0 = blockIdx.x * 128 + wave * 16;  // this wave's 16 rows
  const int nb    = blockIdx.y;                    // 128-col block

  // W staging: thread -> (row, 32B half) of the 128x32 chunk tile
  const int srow  = tid >> 1;
  const int shalf = tid & 1;
  const unsigned short* wgsrc = Wt + (size_t)(nb * 128 + srow) * DDIM + shalf * 16;
  const int sdst = srow * WSTR + shalf * 16;

  // prologue: async-copy chunk 0 into buffer 0
  async_ld16(lds_off(&wlds[0][sdst]),     wgsrc);
  async_ld16(lds_off(&wlds[0][sdst + 8]), wgsrc + 8);

  const unsigned short* xrow = Xb + ((size_t)(mrow0 + nloc)) * DDIM + hi * 8;
  Frag af;
  af.q[0] = *(const u32x4*)(xrow);
  af.q[1] = *(const u32x4*)(xrow + 16);

  const v8f z8 = {0.f,0.f,0.f,0.f,0.f,0.f,0.f,0.f};
  v8f acc[8] = {z8, z8, z8, z8, z8, z8, z8, z8};

  for (int c = 0; c < 16; ++c) {
    const int par = c & 1;
    wait_async_all();                      // my part of chunk c landed
    __syncthreads();                       // everyone's chunk c landed;
                                           // chunk c-1 consumers all done
    if (c < 15) {                          // async-copy next chunk
      const unsigned short* g = wgsrc + (c + 1) * 32;
      async_ld16(lds_off(&wlds[par ^ 1][sdst]),     g);
      async_ld16(lds_off(&wlds[par ^ 1][sdst + 8]), g + 8);
    }
    const int cn = (c < 15) ? c + 1 : c;   // prefetch next A-frag
    Frag afn;
    afn.q[0] = *(const u32x4*)(xrow + cn * 32);
    afn.q[1] = *(const u32x4*)(xrow + cn * 32 + 16);

    #pragma unroll
    for (int nt = 0; nt < 8; ++nt) {
      Frag bf;
      const unsigned short* wp = &wlds[par][(nt * 16 + nloc) * WSTR + hi * 8];
      bf.q[0] = *(const u32x4*)(wp);
      bf.q[1] = *(const u32x4*)(wp + 16);
      acc[nt] = wmma_bf16(af.v, bf.v, acc[nt]);
    }
    af = afn;
  }

  #pragma unroll
  for (int nt = 0; nt < 8; ++nt) {
    const int n = nb * 128 + nt * 16 + nloc;
    const float bv = bias[n];
    if (!transposed) {
      #pragma unroll
      for (int r = 0; r < 8; ++r)
        dst[((size_t)(mrow0 + r + 8 * hi)) * DDIM + n] = f2bf(acc[nt][r] + bv);
    } else {
      union { unsigned short h[8]; u32x4 q; } pk;
      #pragma unroll
      for (int r = 0; r < 8; ++r) pk.h[r] = f2bf(acc[nt][r] + bv);
      const int bb = mrow0 >> 11;
      const int sb = (mrow0 & (SLEN - 1)) + 8 * hi;
      *(u32x4*)(dst + (size_t)bb * DDIM * SLEN + (size_t)n * SLEN + sb) = pk.q;
    }
  }
}

// ---------------------------------------------------------------------------
// Stage 2: flash attention. 1 WG (8 waves) per (batch, 32-row q-tile).
// K tiles (32 keys x 512 d) double-buffered in LDS via async global->LDS
// copies (in flight across the whole compute step). V fragments direct from
// global, issued early, consumed after softmax.
// ---------------------------------------------------------------------------
__global__ __launch_bounds__(256) void attn_fa(const unsigned short* __restrict__ Qb,
                                               const unsigned short* __restrict__ Kb,
                                               const unsigned short* __restrict__ Vt,
                                               float* __restrict__ Out) {
  __shared__ __align__(16) unsigned short klds[2][32 * KSTR]; // 2 x 33.3 KB
  __shared__ __align__(16) float          sbuf[32 * SSTR];    // scores (padded)
  __shared__ __align__(16) unsigned short pbuf[32 * PSTR];    // bf16 P (padded)
  __shared__ float cbuf[32];
  __shared__ float lbuf[32];

  const int tid  = threadIdx.x;
  const int lane = tid & 31;
  const int wave = tid >> 5;
  const int nloc = lane & 15;
  const int hi   = lane >> 4;

  const int b     = blockIdx.x >> 6;
  const int q0    = (blockIdx.x & 63) << 5;
  const int dbase = wave * 64;

  const unsigned short* Kbase = Kb + ((size_t)b * SLEN) * DDIM;
  const unsigned short* Vbase = Vt + ((size_t)b) * DDIM * SLEN;

  // K staging geometry: 2048 16B-chunks per tile, 8 per thread
  const int crow = (tid + 0) >> 6;                  // reused per i with +4 rows
  (void)crow;

  // Q A-fragments: 2 m-tiles x 2 k-chunks, in VGPRs
  Frag qf[2][2];
  #pragma unroll
  for (int mt = 0; mt < 2; ++mt) {
    const unsigned short* qrow =
        Qb + ((size_t)(b * SLEN + q0 + mt * 16 + nloc)) * DDIM + dbase + hi * 8;
    #pragma unroll
    for (int c = 0; c < 2; ++c) {
      qf[mt][c].q[0] = *(const u32x4*)(qrow + c * 32);
      qf[mt][c].q[1] = *(const u32x4*)(qrow + c * 32 + 16);
    }
  }

  const v8f z8 = {0.f,0.f,0.f,0.f,0.f,0.f,0.f,0.f};
  v8f acc[2][4] = {{z8, z8, z8, z8}, {z8, z8, z8, z8}};

  float m_run = -1e30f;
  float l_run = 0.f;

  // prologue: async-copy K tile for j0 = 0 into buffer 0
  #pragma unroll
  for (int i = 0; i < 8; ++i) {
    const int cidx = tid + i * 256;
    const int row  = cidx >> 6;
    const int off  = (cidx & 63) * 8;
    async_ld16(lds_off(&klds[0][row * KSTR + off]),
               Kbase + (size_t)row * DDIM + off);
  }

  for (int j0 = 0; j0 < SLEN; j0 += 32) {
    const int par = (j0 >> 5) & 1;

    // zero score buffer (prev readers finished before prev stats barrier)
    #pragma unroll
    for (int k = 0; k < 5; ++k) {
      const int idx = tid + k * 256;
      if (idx < 32 * SSTR) sbuf[idx] = 0.f;
    }

    wait_async_all();                           // my part of K tile j0 landed
    __syncthreads();                            // all parts landed; buffer
                                                // par^1 fully consumed

    // async-copy next K tile into the other buffer (in flight all step long)
    if (j0 + 32 < SLEN) {
      #pragma unroll
      for (int i = 0; i < 8; ++i) {
        const int cidx = tid + i * 256;
        const int row  = cidx >> 6;
        const int off  = (cidx & 63) * 8;
        async_ld16(lds_off(&klds[par ^ 1][row * KSTR + off]),
                   Kbase + (size_t)(j0 + 32 + row) * DDIM + off);
      }
    }

    // V fragments direct from global, issued now, consumed after softmax
    Frag vf[4];
    #pragma unroll
    for (int nt = 0; nt < 4; ++nt) {
      const unsigned short* vr =
          Vbase + ((size_t)(dbase + nt * 16 + nloc)) * SLEN + j0 + hi * 8;
      vf[nt].q[0] = *(const u32x4*)(vr);
      vf[nt].q[1] = *(const u32x4*)(vr + 16);
    }

    // ---- partial scores over this wave's 64 d-columns (8 WMMAs) ----
    v8f sc[2][2] = {{z8, z8}, {z8, z8}};
    #pragma unroll
    for (int t = 0; t < 2; ++t)
      #pragma unroll
      for (int c = 0; c < 2; ++c) {
        Frag kf;
        const unsigned short* kp =
            &klds[par][(t * 16 + nloc) * KSTR + dbase + c * 32 + hi * 8];
        kf.q[0] = *(const u32x4*)(kp);
        kf.q[1] = *(const u32x4*)(kp + 16);
        sc[0][t] = wmma_bf16(qf[0][c].v, kf.v, sc[0][t]);
        sc[1][t] = wmma_bf16(qf[1][c].v, kf.v, sc[1][t]);
      }

    // ---- cross-wave reduction of the 32x32 score tile ----
    #pragma unroll
    for (int mt = 0; mt < 2; ++mt)
      #pragma unroll
      for (int t = 0; t < 2; ++t)
        #pragma unroll
        for (int r = 0; r < 8; ++r)
          atomicAdd(&sbuf[(mt * 16 + r + 8 * hi) * SSTR + t * 16 + nloc],
                    sc[mt][t][r]);
    __syncthreads();

    // ---- online softmax: wave 0, one q-row per lane ----
    if (wave == 0) {
      const v4f* s4 = (const v4f*)sbuf;
      float rmax = -1e30f;
      #pragma unroll
      for (int jj = 0; jj < 8; ++jj) {
        v4f s = s4[lane * 9 + jj];
        #pragma unroll
        for (int e = 0; e < 4; ++e) rmax = fmaxf(rmax, s[e] * SCALE);
      }
      const float m_new = fmaxf(m_run, rmax);
      const float cexp  = __expf(m_run - m_new);
      float psum = 0.f;
      unsigned int* p32 = (unsigned int*)pbuf;
      #pragma unroll
      for (int jj = 0; jj < 8; ++jj) {
        v4f s = s4[lane * 9 + jj];
        const float p0 = __expf(s[0] * SCALE - m_new);
        const float p1 = __expf(s[1] * SCALE - m_new);
        const float p2 = __expf(s[2] * SCALE - m_new);
        const float p3 = __expf(s[3] * SCALE - m_new);
        psum += (p0 + p1) + (p2 + p3);
        p32[lane * (PSTR / 2) + jj * 2 + 0] =
            (unsigned int)f2bf(p0) | ((unsigned int)f2bf(p1) << 16);
        p32[lane * (PSTR / 2) + jj * 2 + 1] =
            (unsigned int)f2bf(p2) | ((unsigned int)f2bf(p3) << 16);
      }
      l_run = l_run * cexp + psum;
      m_run = m_new;
      cbuf[lane] = cexp;
    }
    __syncthreads();

    // ---- rescale accumulators, then acc += P @ Vt (8 WMMAs) ----
    float cv[2][8];
    #pragma unroll
    for (int mt = 0; mt < 2; ++mt)
      #pragma unroll
      for (int r = 0; r < 8; ++r) cv[mt][r] = cbuf[mt * 16 + r + 8 * hi];
    #pragma unroll
    for (int mt = 0; mt < 2; ++mt)
      #pragma unroll
      for (int nt = 0; nt < 4; ++nt)
        #pragma unroll
        for (int r = 0; r < 8; ++r) acc[mt][nt][r] *= cv[mt][r];

    Frag pf[2];
    #pragma unroll
    for (int mt = 0; mt < 2; ++mt) {
      const unsigned short* pp = pbuf + (mt * 16 + nloc) * PSTR + hi * 8;
      pf[mt].q[0] = *(const u32x4*)(pp);
      pf[mt].q[1] = *(const u32x4*)(pp + 16);
    }

    #pragma unroll
    for (int mt = 0; mt < 2; ++mt)
      #pragma unroll
      for (int nt = 0; nt < 4; ++nt)
        acc[mt][nt] = wmma_bf16(pf[mt].v, vf[nt].v, acc[mt][nt]);
  }

  if (wave == 0) lbuf[lane] = l_run;
  __syncthreads();

  float inv[2][8];
  #pragma unroll
  for (int mt = 0; mt < 2; ++mt)
    #pragma unroll
    for (int r = 0; r < 8; ++r) inv[mt][r] = 1.f / lbuf[mt * 16 + r + 8 * hi];

  #pragma unroll
  for (int mt = 0; mt < 2; ++mt)
    #pragma unroll
    for (int nt = 0; nt < 4; ++nt)
      #pragma unroll
      for (int r = 0; r < 8; ++r)
        Out[((size_t)(b * SLEN + q0 + mt * 16 + r + 8 * hi)) * DDIM +
            dbase + nt * 16 + nloc] = acc[mt][nt][r] * inv[mt][r];
}

// ---------------------------------------------------------------------------
extern "C" void kernel_launch(void* const* d_in, const int* in_sizes, int n_in,
                              void* d_out, int out_size, void* d_ws, size_t ws_size,
                              hipStream_t stream) {
  (void)in_sizes; (void)n_in; (void)out_size; (void)ws_size;

  const float* x  = (const float*)d_in[0];
  const float* Wq = (const float*)d_in[1];
  const float* bq = (const float*)d_in[2];
  const float* Wk = (const float*)d_in[3];
  const float* bk = (const float*)d_in[4];
  const float* Wv = (const float*)d_in[5];
  const float* bv = (const float*)d_in[6];
  float* out = (float*)d_out;

  const size_t NX = (size_t)BATCH * SLEN * DDIM;
  unsigned short* Xb  = (unsigned short*)d_ws;
  unsigned short* Qb  = Xb + NX;
  unsigned short* Kb  = Qb + NX;
  unsigned short* Vt  = Kb + NX;
  unsigned short* Wqt = Vt + NX;
  unsigned short* Wkt = Wqt + (size_t)DDIM * DDIM;
  unsigned short* Wvt = Wkt + (size_t)DDIM * DDIM;

  cvt_x_bf16<<<dim3((unsigned)(NX / 1024)), dim3(256), 0, stream>>>(x, Xb);
  cvt_w_t<<<dim3((DDIM * DDIM) / 256), dim3(256), 0, stream>>>(Wq, Wqt);
  cvt_w_t<<<dim3((DDIM * DDIM) / 256), dim3(256), 0, stream>>>(Wk, Wkt);
  cvt_w_t<<<dim3((DDIM * DDIM) / 256), dim3(256), 0, stream>>>(Wv, Wvt);

  dim3 pg(BATCH * SLEN / 128, DDIM / 128);
  qkv_proj<<<pg, dim3(256), 0, stream>>>(Xb, Wqt, bq, Qb, 0);
  qkv_proj<<<pg, dim3(256), 0, stream>>>(Xb, Wkt, bk, Kb, 0);
  qkv_proj<<<pg, dim3(256), 0, stream>>>(Xb, Wvt, bv, Vt, 1);

  attn_fa<<<dim3(BATCH * SLEN / 32), dim3(256), 0, stream>>>(Qb, Kb, Vt, out);
}
